// FusedEncoderBlock_33827162423668
// MI455X (gfx1250) — matrix operand order repaired and checked by reading, not compiled, not run on hardware
//
#include <hip/hip_runtime.h>
#include <math.h>

// ---------------------------------------------------------------------------
// Fused transformer encoder block for MI455X (gfx1250, wave32, WMMA).
// GEMMs: v_wmma_f32_16x16x32_bf16, double-buffered LDS tiles fed by
// async-to-LDS copies (ASYNCcnt) + Tensor Data Mover (TENSORcnt).
// ---------------------------------------------------------------------------

typedef __bf16 bf16;
typedef __attribute__((ext_vector_type(16))) __bf16 v16bf;
typedef __attribute__((ext_vector_type(8)))  float  v8f;
typedef int          i32x4 __attribute__((vector_size(16)));
typedef int          i32x8 __attribute__((vector_size(32)));
typedef unsigned int u32x4 __attribute__((vector_size(16)));

#define D_MODEL 2048
#define N_HEADS 16
#define D_HEAD  128
#define FF_DIM  8192
#define BATCH   2
#define SEQ     2048
#define M_TOT   (BATCH * SEQ)   // 4096 rows

// ---- CDNA5 feature detection (toolchain-portable) -------------------------
#if defined(__has_builtin)
#if __has_builtin(__builtin_amdgcn_global_load_async_to_lds_b128)
#define HAVE_ASYNC_BUILTIN 1
#endif
#if __has_builtin(__builtin_amdgcn_tensor_load_to_lds)
#define HAVE_TDM 1
#endif
#endif
#ifndef HAVE_ASYNC_BUILTIN
#define HAVE_ASYNC_BUILTIN 0
#endif
#ifndef HAVE_TDM
#define HAVE_TDM 0
#endif

// ---------------------------------------------------------------------------
// WMMA helper:  D = A(16x32 bf16) * B(32x16 bf16) + C(16x16 f32)
// ---------------------------------------------------------------------------
__device__ __forceinline__ v8f wmma_bf16(v16bf a, v16bf b, v8f c) {
  return __builtin_amdgcn_wmma_f32_16x16x32_bf16(
      /*neg_a=*/false, a, /*neg_b=*/false, b,
      /*c_mod=*/(short)0, c, /*reuse_a=*/false, /*reuse_b=*/false);
}

// A-matrix 16x32 bf16 fragment (ISA 7.12.2): lane half h holds K in
// [8h,8h+8) (VGPR0-3) and [16+8h,16+8h+8) (VGPR4-7).
__device__ __forceinline__ v16bf load_a_frag(const bf16* rowbase, int h) {
  v16bf r;
  ((uint4*)&r)[0] = *(const uint4*)(rowbase + 8 * h);
  ((uint4*)&r)[1] = *(const uint4*)(rowbase + 16 + 8 * h);
  return r;
}
// B-matrix 32x16 bf16 fragment: lane half h holds K in [8h,8h+8) and
// [16+8h,16+8h+8) (same K order as the A fragment).
__device__ __forceinline__ v16bf load_b_frag(const bf16* rowbase, int h) {
  v16bf r;
  ((uint4*)&r)[0] = *(const uint4*)(rowbase + 8 * h);
  ((uint4*)&r)[1] = *(const uint4*)(rowbase + 16 + 8 * h);
  return r;
}

// ---------------------------------------------------------------------------
// Async global -> LDS 16-byte copy (ASYNCcnt path)
// ---------------------------------------------------------------------------
__device__ __forceinline__ void async_copy16(const bf16* g, bf16* l) {
#if HAVE_ASYNC_BUILTIN
  __builtin_amdgcn_global_load_async_to_lds_b128(
      (__attribute__((address_space(1))) i32x4*)(
          __attribute__((address_space(1))) void*)(void*)g,
      (__attribute__((address_space(3))) i32x4*)(
          __attribute__((address_space(3))) void*)(void*)l,
      0, 0);
#else
  const unsigned loff =
      (unsigned)(unsigned long long)(__attribute__((address_space(3))) void*)l;
  asm volatile("global_load_async_to_lds_b128 %0, %1, off" ::"v"(loff), "v"(g)
               : "memory");
#endif
}

#if HAVE_TDM
// TDM: DMA a 128x32 bf16 tile (row stride = ld elems) from global into LDS.
// D# group0/group1 packed per ISA 8.3/8.4; 2-D tensor, groups 2/3 zero.
// This toolchain's builtin takes 6 args (g0, g1, g2, g3, g4, cpol).
__device__ __forceinline__ void tdm_load_tile_128x32(const bf16* gtile,
                                                     const bf16* ldsdst,
                                                     int ld) {
  const unsigned lds_addr = (unsigned)(unsigned long long)(
      __attribute__((address_space(3))) void*)(void*)(bf16*)ldsdst;
  const unsigned long long ga = (unsigned long long)gtile;
  u32x4 g0;
  g0[0] = 1u;                                            // count=1
  g0[1] = lds_addr;                                      // lds_addr
  g0[2] = (unsigned)ga;                                  // global_addr lo
  g0[3] = (unsigned)((ga >> 32) & 0x01FFFFFFu) | 0x80000000u;  // hi | type=2
  i32x8 g1;
  g1[0] = 0x00010000;                                    // data_size=1 (2B)
  g1[1] = (int)(((unsigned)ld & 0xFFFFu) << 16);         // tensor_dim0 lo16
  g1[2] = (int)(((unsigned)ld >> 16) | (128u << 16));    // dim0 hi16|dim1=128
  g1[3] = (int)(32u << 16);                              // tile_dim0=32
  g1[4] = 128;                                           // tile_dim1=128
  g1[5] = ld;                                            // dim0_stride lo32
  g1[6] = 0;
  g1[7] = 0;
  const i32x4 z4 = {0, 0, 0, 0};
  const i32x8 z8 = {0, 0, 0, 0, 0, 0, 0, 0};
  __builtin_amdgcn_tensor_load_to_lds(g0, g1, z4, z4, z8, 0);
}
#endif

#if HAVE_TDM
#define ASYNC_PENDING_IMM "0x2"  // 2 per-thread async ops/stage (A tile only)
#else
#define ASYNC_PENDING_IMM "0x4"  // 4 per-thread async ops/stage (A + W tiles)
#endif
__device__ __forceinline__ void wait_async(bool pending) {
  if (pending)
    asm volatile("s_wait_asynccnt " ASYNC_PENDING_IMM ::: "memory");
  else
    asm volatile("s_wait_asynccnt 0x0" ::: "memory");
}

// ---------------------------------------------------------------------------
// fp32 -> bf16 conversion (weights; memory-bound pre-pass)
// ---------------------------------------------------------------------------
__global__ void convert_f32_bf16_kernel(const float* __restrict__ in,
                                        bf16* __restrict__ out, long n) {
  long i = (long)blockIdx.x * blockDim.x + threadIdx.x;
  long stride = (long)gridDim.x * blockDim.x;
  for (; i < n; i += stride) out[i] = (bf16)in[i];
}

// ---------------------------------------------------------------------------
// LayerNorm over D_MODEL, one block per row, bf16 output
// ---------------------------------------------------------------------------
__global__ __launch_bounds__(256) void layernorm_kernel(
    const float* __restrict__ x, const float* __restrict__ g,
    const float* __restrict__ beta, bf16* __restrict__ out) {
  const int row = blockIdx.x;
  const int tid = threadIdx.x;
  const float* xr = x + (size_t)row * D_MODEL;
  float s = 0.f, ss = 0.f;
  for (int i = tid; i < D_MODEL; i += 256) {
    float v = xr[i];
    s += v;
    ss += v * v;
  }
  __shared__ float rs[256], rss[256];
  rs[tid] = s; rss[tid] = ss;
  __syncthreads();
  for (int off = 128; off; off >>= 1) {
    if (tid < off) { rs[tid] += rs[tid + off]; rss[tid] += rss[tid + off]; }
    __syncthreads();
  }
  const float mu  = rs[0] * (1.f / D_MODEL);
  const float var = rss[0] * (1.f / D_MODEL) - mu * mu;
  const float inv = rsqrtf(var + 1e-5f);
  bf16* orow = out + (size_t)row * D_MODEL;
  for (int i = tid; i < D_MODEL; i += 256)
    orow[i] = (bf16)((xr[i] - mu) * inv * g[i] + beta[i]);
}

// ---------------------------------------------------------------------------
// RoPE applied in-place to q and k halves of the packed qkv bf16 buffer.
// ---------------------------------------------------------------------------
__global__ void rope_kernel(bf16* __restrict__ qkv) {
  const long total = (long)M_TOT * 2 * N_HEADS * 64;
  long i = (long)blockIdx.x * blockDim.x + threadIdx.x;
  if (i >= total) return;
  const int d  = (int)(i % 64);        i /= 64;
  const int h  = (int)(i % N_HEADS);   i /= N_HEADS;
  const int qk = (int)(i % 2);         i /= 2;
  const long m = i;                    // global row (b*SEQ + s)
  const int sq = (int)(m % SEQ);
  const float inv = __powf(10000.f, -(float)d * (1.f / 64.f));
  const float ang = (float)sq * inv;
  const float c = __cosf(ang), s = __sinf(ang);
  bf16* base = qkv + (size_t)m * (3 * D_MODEL) + (size_t)qk * D_MODEL + h * D_HEAD;
  const float x1 = (float)base[d];
  const float x2 = (float)base[d + 64];
  base[d]      = (bf16)(x1 * c - x2 * s);
  base[d + 64] = (bf16)(x2 * c + x1 * s);
}

// ---------------------------------------------------------------------------
// Tiled WMMA GEMM:  out[M,N] = A[M,K] @ W[N,K]^T  (+ epilogue)
//   EPI 0: store bf16 (no bias)          -> QKV
//   EPI 1: +bias +resid(f32), store f32  -> proj & FF2
//   EPI 2: +bias, exact GELU, store bf16 -> FF1
// 256 thr (8 waves, 2x4), tile 128x128x32, double-buffered LDS stages:
// A tile via async-to-LDS (ASYNCcnt), W tile via TDM (TENSORcnt).
// ---------------------------------------------------------------------------
template <int EPI>
__global__ __launch_bounds__(256) void gemm_bf16_kernel(
    const bf16* __restrict__ A, const bf16* __restrict__ W,
    int M, int N, int K,
    const float* __restrict__ bias, const float* __restrict__ resid,
    float* __restrict__ outF, bf16* __restrict__ outB) {
  __shared__ __align__(32) bf16 As[2][128 * 32];
  __shared__ __align__(32) bf16 Ws[2][128 * 32];
  const int tid  = threadIdx.x;
  const int lane = tid & 31, wave = tid >> 5;
  const int wm = wave >> 2, wn = wave & 3;  // 2 x 4 waves
  const int ln = lane & 15, lh = lane >> 4;
  const int m0 = blockIdx.y * 128, n0 = blockIdx.x * 128;

  v8f acc[4][2] = {};

  auto stage = [&](int b, int k0) {
    // A tile: 128x32 bf16, 2 x 16B async copies per thread.
#pragma unroll
    for (int i = 0; i < 2; ++i) {
      const int idx = tid + i * 256;
      const int r = idx >> 2, c = (idx & 3) * 8;
      async_copy16(A + (size_t)(m0 + r) * K + k0 + c, &As[b][r * 32 + c]);
#if !HAVE_TDM
      async_copy16(W + (size_t)(n0 + r) * K + k0 + c, &Ws[b][r * 32 + c]);
#endif
    }
#if HAVE_TDM
    if (wave == 0) tdm_load_tile_128x32(W + (size_t)n0 * K + k0, &Ws[b][0], K);
#endif
  };

  stage(0, 0);
  int buf = 0;
  for (int k0 = 0; k0 < K; k0 += 32) {
    const bool nxt = (k0 + 32) < K;
    if (nxt) stage(buf ^ 1, k0 + 32);     // keep one stage in flight
    wait_async(nxt);                      // current A tile landed
#if HAVE_TDM
    if (wave == 0) {                      // current W tile landed
      if (nxt) __builtin_amdgcn_s_wait_tensorcnt(1);
      else     __builtin_amdgcn_s_wait_tensorcnt(0);
    }
#endif
    __syncthreads();

    v16bf af[4], wf[2];
#pragma unroll
    for (int mt = 0; mt < 4; ++mt)
      af[mt] = load_a_frag(&As[buf][(wm * 64 + mt * 16 + ln) * 32], lh);
#pragma unroll
    for (int nt = 0; nt < 2; ++nt)
      wf[nt] = load_b_frag(&Ws[buf][(wn * 32 + nt * 16 + ln) * 32], lh);
#pragma unroll
    for (int mt = 0; mt < 4; ++mt)
#pragma unroll
      for (int nt = 0; nt < 2; ++nt)
        acc[mt][nt] = wmma_bf16(af[mt], wf[nt], acc[mt][nt]);

    __syncthreads();                      // tile consumed; buf may be restaged
    buf ^= 1;
  }

  // Epilogue (C/D layout: VGPR r -> row r + 8*lh, col = lane%16)
#pragma unroll
  for (int mt = 0; mt < 4; ++mt) {
#pragma unroll
    for (int nt = 0; nt < 2; ++nt) {
#pragma unroll
      for (int r = 0; r < 8; ++r) {
        const int m = m0 + wm * 64 + mt * 16 + r + 8 * lh;
        const int n = n0 + wn * 32 + nt * 16 + ln;
        const size_t o = (size_t)m * N + n;
        float v = acc[mt][nt][r];
        if (EPI == 0) {
          outB[o] = (bf16)v;
        } else if (EPI == 1) {
          v += bias[n] + resid[o];
          outF[o] = v;
        } else {  // EPI == 2: exact GELU
          v += bias[n];
          v = 0.5f * v * (1.f + erff(v * 0.70710678118654752f));
          outB[o] = (bf16)v;
        }
      }
    }
  }
}

// ---------------------------------------------------------------------------
// Causal flash attention. Grid: (SEQ/128, BATCH*N_HEADS), 256 threads.
// Each wave independently owns 16 q rows (online softmax in C/D layout);
// per step: 32 keys -> 8 QK^T WMMAs + 8 PV WMMAs.
// ---------------------------------------------------------------------------
__global__ __launch_bounds__(256) void flash_attn_kernel(
    const bf16* __restrict__ qkv, bf16* __restrict__ O) {
  __shared__ __align__(32) bf16 vt[128 * 32];       // V^T tile: [d][k]
  __shared__ __align__(32) bf16 pbuf[8][16 * 32];   // per-wave P: [m][k]

  const int bh = blockIdx.y;
  const int b = bh / N_HEADS, h = bh % N_HEADS;
  const int q0 = blockIdx.x * 128;
  const int tid = threadIdx.x;
  const int lane = tid & 31, wave = tid >> 5;
  const int ln = lane & 15, lh = lane >> 4;

  const size_t rowstride = 3 * D_MODEL;
  const bf16* qb = qkv + (size_t)b * SEQ * rowstride + (size_t)h * D_HEAD;
  const bf16* kb = qb + D_MODEL;
  const bf16* vb = qb + 2 * D_MODEL;

  const int sq_base = q0 + wave * 16;

  v16bf qf[4];
  {
    const int qrow = sq_base + ln;
#pragma unroll
    for (int d4 = 0; d4 < 4; ++d4)
      qf[d4] = load_a_frag(qb + (size_t)qrow * rowstride + d4 * 32, lh);
  }

  v8f o[8] = {};
  float mrow[8], lrow[8];
#pragma unroll
  for (int r = 0; r < 8; ++r) { mrow[r] = -1e30f; lrow[r] = 0.f; }

  const float scale = 0.08838834764831845f;  // 1/sqrt(128)
  const int kmax = q0 + 128;

  for (int kt = 0; kt < kmax; kt += 32) {
    __syncthreads();  // previous iteration's vt readers done
#pragma unroll
    for (int it = 0; it < 2; ++it) {
      const int c = tid + it * 256;          // 512 chunks of 8 elems
      const int kk = c >> 4, d0 = (c & 15) * 8;
      uint4 u = *(const uint4*)(vb + (size_t)(kt + kk) * rowstride + d0);
      const bf16* pv = (const bf16*)&u;
#pragma unroll
      for (int j = 0; j < 8; ++j) vt[(d0 + j) * 32 + kk] = pv[j];
    }
    __syncthreads();

    v8f s0 = {}, s1 = {};
#pragma unroll
    for (int d4 = 0; d4 < 4; ++d4) {
      const int krow = kt + ln;
      v16bf kf0 = load_b_frag(kb + (size_t)krow * rowstride + d4 * 32, lh);
      v16bf kf1 = load_b_frag(kb + (size_t)(krow + 16) * rowstride + d4 * 32, lh);
      s0 = wmma_bf16(qf[d4], kf0, s0);
      s1 = wmma_bf16(qf[d4], kf1, s1);
    }

#pragma unroll
    for (int r = 0; r < 8; ++r) {
      const int sq = sq_base + r + 8 * lh;
      const int sk0 = kt + ln, sk1 = sk0 + 16;
      float a = s0[r] * scale; if (sk0 > sq) a = -1e30f;
      float bsc = s1[r] * scale; if (sk1 > sq) bsc = -1e30f;
      float mx = fmaxf(a, bsc);
#pragma unroll
      for (int off = 1; off < 16; off <<= 1)
        mx = fmaxf(mx, __shfl_xor(mx, off, 16));
      const float newm = fmaxf(mrow[r], mx);
      const float sc = __expf(mrow[r] - newm);
      const float p0 = __expf(a - newm);
      const float p1 = __expf(bsc - newm);
      float ps = p0 + p1;
#pragma unroll
      for (int off = 1; off < 16; off <<= 1) ps += __shfl_xor(ps, off, 16);
      lrow[r] = lrow[r] * sc + ps;
      mrow[r] = newm;
#pragma unroll
      for (int nt = 0; nt < 8; ++nt) o[nt][r] *= sc;
      const int mloc = r + 8 * lh;
      pbuf[wave][mloc * 32 + ln]      = (bf16)p0;
      pbuf[wave][mloc * 32 + ln + 16] = (bf16)p1;
    }

    asm volatile("s_wait_dscnt 0x0" ::: "memory");

    const v16bf pf = load_a_frag(&pbuf[wave][ln * 32], lh);
#pragma unroll
    for (int nt = 0; nt < 8; ++nt) {
      const v16bf vf = load_b_frag(vt + (nt * 16 + ln) * 32, lh);
      o[nt] = wmma_bf16(pf, vf, o[nt]);
    }
  }

#pragma unroll
  for (int nt = 0; nt < 8; ++nt) {
#pragma unroll
    for (int r = 0; r < 8; ++r) {
      const int sq = sq_base + r + 8 * lh;
      const int d = nt * 16 + ln;
      const float val = o[nt][r] / lrow[r];
      O[((size_t)b * SEQ + sq) * D_MODEL + h * D_HEAD + d] = (bf16)val;
    }
  }
}

// ---------------------------------------------------------------------------
// Host-side orchestration
// ---------------------------------------------------------------------------
extern "C" void kernel_launch(void* const* d_in, const int* in_sizes, int n_in,
                              void* d_out, int out_size, void* d_ws,
                              size_t ws_size, hipStream_t stream) {
  (void)in_sizes; (void)n_in; (void)out_size; (void)ws_size;
  const float* x      = (const float*)d_in[0];
  const float* w_qkv  = (const float*)d_in[1];
  const float* w_proj = (const float*)d_in[2];
  const float* b_proj = (const float*)d_in[3];
  const float* w_ff1  = (const float*)d_in[4];
  const float* b_ff1  = (const float*)d_in[5];
  const float* w_ff2  = (const float*)d_in[6];
  const float* b_ff2  = (const float*)d_in[7];
  const float* g1     = (const float*)d_in[8];
  const float* beta1  = (const float*)d_in[9];
  const float* g2     = (const float*)d_in[10];
  const float* beta2  = (const float*)d_in[11];
  float* out = (float*)d_out;

  char* ws = (char*)d_ws;
  size_t off = 0;
  auto take = [&](size_t bytes) -> char* {
    char* p = ws + off;
    off = (off + bytes + 255) & ~(size_t)255;
    return p;
  };
  bf16* h1    = (bf16*)take((size_t)M_TOT * D_MODEL * 2);
  bf16* qkv   = (bf16*)take((size_t)M_TOT * 3 * D_MODEL * 2);
  bf16* wqkvB = (bf16*)take((size_t)3 * D_MODEL * D_MODEL * 2);
  bf16* wprjB = (bf16*)take((size_t)D_MODEL * D_MODEL * 2);
  bf16* wff1B = (bf16*)take((size_t)FF_DIM * D_MODEL * 2);
  bf16* wff2B = (bf16*)take((size_t)D_MODEL * FF_DIM * 2);
  bf16* attnO = (bf16*)take((size_t)M_TOT * D_MODEL * 2);
  float* x1   = (float*)take((size_t)M_TOT * D_MODEL * 4);
  bf16* h2    = (bf16*)take((size_t)M_TOT * D_MODEL * 2);
  bf16* act   = (bf16*)take((size_t)M_TOT * FF_DIM * 2);

  convert_f32_bf16_kernel<<<8192, 256, 0, stream>>>(w_qkv, wqkvB, (long)3 * D_MODEL * D_MODEL);
  convert_f32_bf16_kernel<<<4096, 256, 0, stream>>>(w_proj, wprjB, (long)D_MODEL * D_MODEL);
  convert_f32_bf16_kernel<<<8192, 256, 0, stream>>>(w_ff1, wff1B, (long)FF_DIM * D_MODEL);
  convert_f32_bf16_kernel<<<8192, 256, 0, stream>>>(w_ff2, wff2B, (long)D_MODEL * FF_DIM);

  layernorm_kernel<<<M_TOT, 256, 0, stream>>>(x, g1, beta1, h1);

  gemm_bf16_kernel<0><<<dim3(3 * D_MODEL / 128, M_TOT / 128), 256, 0, stream>>>(
      h1, wqkvB, M_TOT, 3 * D_MODEL, D_MODEL, nullptr, nullptr, nullptr, qkv);

  {
    const long total = (long)M_TOT * 2 * N_HEADS * 64;
    rope_kernel<<<(unsigned)((total + 255) / 256), 256, 0, stream>>>(qkv);
  }

  flash_attn_kernel<<<dim3(SEQ / 128, BATCH * N_HEADS), 256, 0, stream>>>(qkv, attnO);

  gemm_bf16_kernel<1><<<dim3(D_MODEL / 128, M_TOT / 128), 256, 0, stream>>>(
      attnO, wprjB, M_TOT, D_MODEL, D_MODEL, b_proj, x, x1, nullptr);

  layernorm_kernel<<<M_TOT, 256, 0, stream>>>(x1, g2, beta2, h2);

  gemm_bf16_kernel<2><<<dim3(FF_DIM / 128, M_TOT / 128), 256, 0, stream>>>(
      h2, wff1B, M_TOT, FF_DIM, D_MODEL, b_ff1, nullptr, nullptr, act);

  gemm_bf16_kernel<1><<<dim3(D_MODEL / 128, M_TOT / 128), 256, 0, stream>>>(
      act, wff2B, M_TOT, D_MODEL, FF_DIM, b_ff2, x1, out, nullptr);
}
